// SmoothQuantLinear_7524782702654
// MI455X (gfx1250) — compile-verified, matrix operand order checked by reading, not run on hardware
//
#include <hip/hip_runtime.h>

typedef _Float16 half_t;
typedef __attribute__((ext_vector_type(16))) _Float16 v16h;
typedef __attribute__((ext_vector_type(8)))  _Float16 h8;
typedef __attribute__((ext_vector_type(2)))  _Float16 h2;
typedef __attribute__((ext_vector_type(8)))  float    v8f;
typedef __attribute__((ext_vector_type(4)))  int      i32x4;

#define BM 128
#define BN 128
#define BK 64
#define GROUPSZ 128
#define LDS_STRIDE 72   // BK + 8 halves padding (144B rows, 16B aligned)
#define KMAX 4096       // rsmooth LDS table capacity (K == 4096 here)

#if __has_builtin(__builtin_amdgcn_global_load_async_to_lds_b128)
#define USE_ASYNC_A 1
typedef __attribute__((address_space(1))) i32x4 g_i32x4;   // global (AS1)
typedef __attribute__((address_space(3))) i32x4 l_i32x4;   // LDS    (AS3)
#else
#define USE_ASYNC_A 0
#endif

union V16   { v16h v; h8 h[2]; };
union U32H2 { unsigned int u; h2 h; };

static __device__ inline _Float16 h_rcp(_Float16 s) {
#if __has_builtin(__builtin_amdgcn_rcph)
    return __builtin_amdgcn_rcph(s);
#else
    return (_Float16)(1.0f / (float)s);
#endif
}

static __device__ inline void wait_async_zero() {
#if USE_ASYNC_A
#if __has_builtin(__builtin_amdgcn_s_wait_asynccnt)
    __builtin_amdgcn_s_wait_asynccnt(0);
#else
    asm volatile("s_wait_asynccnt 0x0" ::: "memory");
#endif
#endif
}

__global__ __launch_bounds__(256)
void smoothquant_wmma_kernel(const half_t* __restrict__ x,
                             const int*    __restrict__ pw,
                             const half_t* __restrict__ wscale,
                             const half_t* __restrict__ wzero,
                             const half_t* __restrict__ smooth,
                             const half_t* __restrict__ bias,
                             half_t* __restrict__ out,
                             int M, int N, int K)
{
    __shared__ half_t As[2][BM * LDS_STRIDE];
    __shared__ half_t Bs[2][BN * LDS_STRIDE];
    __shared__ half_t rsmooth[KMAX];

    const int tid  = threadIdx.x;
    const int lane = tid & 31;
    const int wave = tid >> 5;
    const int wm   = wave >> 2;   // 0..1  -> 64-row slab
    const int wn   = wave & 3;    // 0..3  -> 32-col slab

    const int bm0 = blockIdx.y * BM;
    const int bn0 = blockIdx.x * BN;

    // ---- one-time: reciprocal-of-smooth table in LDS (v_rcp_f16) ----
    for (int k = tid * 16; k < K; k += 256 * 16) {
        h8 s0 = *(const h8*)(smooth + k);
        h8 s1 = *(const h8*)(smooth + k + 8);
        h8 r0, r1;
        #pragma unroll
        for (int j = 0; j < 8; ++j) { r0[j] = h_rcp(s0[j]); r1[j] = h_rcp(s1[j]); }
        *(h8*)&rsmooth[k]     = r0;
        *(h8*)&rsmooth[k + 8] = r1;
    }
    __syncthreads();

    // ---- cooperative loader mapping: 2 threads per tile row ----
    const int  lrow = tid >> 1;        // 0..127
    const int  lsel = tid & 1;         // which half of BK
    const int  lkc  = lsel * 32;       // half-element offset within BK
    const int  gm   = bm0 + lrow;
    const int  gn   = bn0 + lrow;
    const long xrow = (long)gm * K;
    const long wrow = (long)gn * (K >> 1);

    v8f acc[4][2];
    #pragma unroll
    for (int i = 0; i < 4; ++i)
        #pragma unroll
        for (int j = 0; j < 2; ++j)
            acc[i][j] = (v8f)0.f;

    // staging registers (B only; A goes global->LDS via async DMA)
    i32x4 qv[4];
    float sc = 0.f, zp = 0.f;
#if !USE_ASYNC_A
    h8 xv[4];
#endif

    // A tile: raw copy of x (smoothing folded into B dequant)
    auto issueA = [&](int buf, int kt) {
        const half_t* xp = x + xrow + kt + lkc;
        half_t*       ap = &As[buf][lrow * LDS_STRIDE + lkc];
#if USE_ASYNC_A
        __builtin_amdgcn_global_load_async_to_lds_b128(
            (g_i32x4*)(xp),      (l_i32x4*)(ap),      0, 0);
        __builtin_amdgcn_global_load_async_to_lds_b128(
            (g_i32x4*)(xp + 8),  (l_i32x4*)(ap + 8),  0, 0);
        __builtin_amdgcn_global_load_async_to_lds_b128(
            (g_i32x4*)(xp + 16), (l_i32x4*)(ap + 16), 0, 0);
        __builtin_amdgcn_global_load_async_to_lds_b128(
            (g_i32x4*)(xp + 24), (l_i32x4*)(ap + 24), 0, 0);
#else
        #pragma unroll
        for (int j = 0; j < 4; ++j) xv[j] = *(const h8*)(xp + 8 * j);
#endif
        __builtin_prefetch(xp + BK, 0, 1);   // global_prefetch_b8
    };

#if !USE_ASYNC_A
    auto flushA = [&](int buf) {
        half_t* ap = &As[buf][lrow * LDS_STRIDE + lkc];
        #pragma unroll
        for (int j = 0; j < 4; ++j) *(h8*)(ap + 8 * j) = xv[j];
    };
#endif

    auto ldglobalB = [&](int kt) {
        const int* qp = pw + wrow + (kt >> 1) + lsel * 16;
        #pragma unroll
        for (int j = 0; j < 4; ++j) qv[j] = *(const i32x4*)(qp + 4 * j);
        const int g = kt / GROUPSZ;   // BK=64 divides GROUPSZ=128, aligned
        sc = (float)wscale[(long)g * N + gn];
        zp = (float)wzero [(long)g * N + gn];
        __builtin_prefetch(qp + (BK >> 1), 0, 1);
    };

    auto stldsB = [&](int buf, int kt) {
        // int4 dequant via 0x6400 exponent trick, smoothing reciprocal folded in:
        // w[k][n] = ((q-8) - zp)*sc * (1/s[k]) = (as_half(0x6400|q)*sc - (1032+zp)*sc) * rs[k]
        const _Float16 sch  = (_Float16)sc;
        const _Float16 offh = (_Float16)((1032.0f + zp) * sc);
        const h2 sc2  = { sch,  sch  };
        const h2 off2 = { offh, offh };
        half_t* bp = &Bs[buf][lrow * LDS_STRIDE + lkc];
        #pragma unroll
        for (int j = 0; j < 4; ++j) {
            h8 rs = *(const h8*)&rsmooth[kt + lkc + 8 * j];
            h8 w;
            #pragma unroll
            for (int e = 0; e < 4; ++e) {
                const unsigned int b = (unsigned int)qv[j][e];
                U32H2 t;
                t.u = 0x64006400u | (b & 0xFu) | ((b & 0xF0u) << 12);
                h2 rsp = { rs[2 * e], rs[2 * e + 1] };
                h2 p = (t.h * sc2 - off2) * rsp;   // v_pk_fma_f16 + v_pk_mul_f16
                w[2 * e]     = p[0];               // even k = low nibble
                w[2 * e + 1] = p[1];               // odd  k = high nibble
            }
            *(h8*)(bp + 8 * j) = w;
        }
    };

    auto compute = [&](int buf) {
        const int r  = lane & 15;
        const int hs = (lane >> 4) * 8;   // K sub-offset per ISA A/B layout
        #pragma unroll
        for (int kk = 0; kk < BK; kk += 32) {
            V16 afr[4], bfr[2];
            #pragma unroll
            for (int fm = 0; fm < 4; ++fm) {
                const half_t* p = &As[buf][(wm * 64 + fm * 16 + r) * LDS_STRIDE + kk + hs];
                afr[fm].h[0] = *(const h8*)(p);        // K = hs .. hs+7
                afr[fm].h[1] = *(const h8*)(p + 16);   // K = hs+16 .. hs+23
            }
            #pragma unroll
            for (int fn = 0; fn < 2; ++fn) {
                const half_t* p = &Bs[buf][(wn * 32 + fn * 16 + r) * LDS_STRIDE + kk + hs];
                bfr[fn].h[0] = *(const h8*)(p);
                bfr[fn].h[1] = *(const h8*)(p + 16);
            }
            #pragma unroll
            for (int fm = 0; fm < 4; ++fm)
                #pragma unroll
                for (int fn = 0; fn < 2; ++fn)
                    acc[fm][fn] = __builtin_amdgcn_wmma_f32_16x16x32_f16(
                        false, afr[fm].v, false, bfr[fn].v,
                        (short)0, acc[fm][fn], false, false);
        }
    };

    // ---- software-pipelined main loop, double-buffered LDS ----
    issueA(0, 0);
    ldglobalB(0);
    stldsB(0, 0);
#if !USE_ASYNC_A
    flushA(0);
#endif
    wait_async_zero();      // A DMA for stage 0 landed (this wave's rows)
    __syncthreads();        // all waves' stage-0 writes visible

    const int NSTEP = K / BK;
    for (int i = 0; i < NSTEP; ++i) {
        const int cur = i & 1;
        const bool more = (i + 1) < NSTEP;
        if (more) { issueA(cur ^ 1, (i + 1) * BK); ldglobalB((i + 1) * BK); }
        compute(cur);
        if (more) {
            stldsB(cur ^ 1, (i + 1) * BK);
#if !USE_ASYNC_A
            flushA(cur ^ 1);
#endif
            wait_async_zero();   // this wave's next-stage A DMA complete
        }
        __syncthreads();
    }

    // ---- epilogue: bias add + fp16 store; C/D VGPR layout mapping ----
    const int r  = lane & 15;
    const int mh = (lane >> 4) * 8;
    #pragma unroll
    for (int fm = 0; fm < 4; ++fm) {
        #pragma unroll
        for (int fn = 0; fn < 2; ++fn) {
            const int   n  = bn0 + wn * 32 + fn * 16 + r;
            const float bv = (float)bias[n];
            const int   mb = bm0 + wm * 64 + fm * 16 + mh;
            #pragma unroll
            for (int i = 0; i < 8; ++i)
                out[(long)(mb + i) * N + n] = (half_t)(acc[fm][fn][i] + bv);
        }
    }
}

extern "C" void kernel_launch(void* const* d_in, const int* in_sizes, int n_in,
                              void* d_out, int out_size, void* d_ws, size_t ws_size,
                              hipStream_t stream) {
    const half_t* x      = (const half_t*)d_in[0];
    const int*    pw     = (const int*)   d_in[1];
    const half_t* wscale = (const half_t*)d_in[2];
    const half_t* wzero  = (const half_t*)d_in[3];
    const half_t* smooth = (const half_t*)d_in[4];
    const half_t* bias   = (const half_t*)d_in[5];
    half_t*       out    = (half_t*)d_out;

    const int K = in_sizes[4];          // smooth_scale has K elements
    const int N = in_sizes[5];          // bias has N elements
    const int M = in_sizes[0] / K;      // x is M*K

    dim3 grid((N + BN - 1) / BN, (M + BM - 1) / BM);
    smoothquant_wmma_kernel<<<grid, 256, 0, stream>>>(
        x, pw, wscale, wzero, smooth, bias, out, M, N, K);
}
